// NeuralFingerprint_49417893708435
// MI455X (gfx1250) — matrix operand order repaired and checked by reading, not compiled, not run on hardware
//
#include <hip/hip_runtime.h>
#include <stdint.h>

typedef __attribute__((ext_vector_type(16))) _Float16 v16h;
typedef __attribute__((ext_vector_type(8)))  _Float16 v8h;
typedef __attribute__((ext_vector_type(8)))  float    v8f;

#define FEAT   128
#define FP_LEN 2048
#define RADIUS 3

static __device__ __forceinline__ void atomAddF(float* p, float v) {
    unsafeAtomicAdd(p, v);   // hardware global_atomic_add_f32 on gfx1250
}

// ---------------------------------------------------------------------------
// Pack a row-major fp32 weight [K x N] into f16 WMMA B-fragments.
// Fragment (nt, kt) stored at ((nt*KT + kt)*32 + lane)*16 halves:
//   lane l<16 : column n = nt*16+l,      k = kt*32 + 0..7, 16..23
//   lane l>=16: column n = nt*16+(l-16), k = kt*32 + 8..15, 24..31
// ---------------------------------------------------------------------------
__global__ void pack_weights(const float* __restrict__ W, _Float16* __restrict__ out,
                             int K, int N) {
    int t = blockIdx.x * blockDim.x + threadIdx.x;
    int KT = K >> 5;
    int NT = N >> 4;
    int total = KT * NT * 32;
    if (t >= total) return;
    int lane = t & 31;
    int frag = t >> 5;
    int kt = frag % KT;
    int nt = frag / KT;
    int n  = nt * 16 + (lane & 15);
    int k0 = kt * 32 + ((lane >> 4) << 3);
    _Float16* o = out + (size_t)t * 16;
#pragma unroll
    for (int i = 0; i < 16; ++i)
        o[i] = (_Float16)W[(size_t)(k0 + i + (i >= 8 ? 8 : 0)) * N + n];
}

__global__ void zero_kernel(float* __restrict__ p, int n) {
    int t = blockIdx.x * blockDim.x + threadIdx.x;
    if (t < n) p[t] = 0.0f;
}

__global__ void copy_f4(const float4* __restrict__ src, float4* __restrict__ dst, int n4) {
    int t = blockIdx.x * blockDim.x + threadIdx.x;
    if (t < n4) dst[t] = src[t];
}

// ---------------------------------------------------------------------------
// Edge scatter-add: agg[dst] += x[src].  One wave per edge, lane covers 4 feats.
// ---------------------------------------------------------------------------
__global__ void scatter_edges(const float* __restrict__ x,
                              const int* __restrict__ esrc,
                              const int* __restrict__ edst,
                              float* __restrict__ agg, int nEdges) {
    int t = blockIdx.x * blockDim.x + threadIdx.x;
    int e = t >> 5;
    if (e >= nEdges) return;
    int lane = t & 31;
    int s = esrc[e];
    int d = edst[e];
    float4 v = *(const float4*)(x + (size_t)s * FEAT + lane * 4);
    float* a = agg + (size_t)d * FEAT + lane * 4;
    atomAddF(a + 0, v.x);
    atomAddF(a + 1, v.y);
    atomAddF(a + 2, v.z);
    atomAddF(a + 3, v.w);
}

// ---------------------------------------------------------------------------
// Cooperatively stage the block's four A-fragments (16x32 f16 each, MxK) into
// LDS in exact v_wmma_f32_16x16x32_f16 A layout, from a row-major fp32 matrix
// with row pitch FEAT.  One 8-half chunk per thread (256 threads = 2048 halves).
// Fragment layout per lane:
//   lane l<16 : row M=l,    halves 0..7 = K 0..7,  halves 8..15 = K 16..23
//   lane l>=16: row M=l-16, halves 0..7 = K 8..15, halves 8..15 = K 24..31
// ---------------------------------------------------------------------------
static __device__ __forceinline__ void stage_a_frags(const float* __restrict__ A,
                                                     int blockM, int tid,
                                                     _Float16 (*aF)[32][16]) {
    int c    = tid;          // 0..255
    int kt   = c >> 6;       // 4 fragments
    int rem  = c & 63;
    int lane = rem >> 1;     // fragment lane 0..31
    int hs   = rem & 1;      // low/high 8-half chunk
    int l15  = lane & 15;
    int k0   = kt * 32 + ((lane >= 16) ? (hs ? 24 : 8) : (hs ? 16 : 0));
    const float* p = A + (size_t)(blockM * 16 + l15) * FEAT + k0;
    float4 f0 = *(const float4*)(p);
    float4 f1 = *(const float4*)(p + 4);
    v8h h;
    h[0] = (_Float16)f0.x; h[1] = (_Float16)f0.y; h[2] = (_Float16)f0.z; h[3] = (_Float16)f0.w;
    h[4] = (_Float16)f1.x; h[5] = (_Float16)f1.y; h[6] = (_Float16)f1.z; h[7] = (_Float16)f1.w;
    *(v8h*)&aF[kt][lane][hs * 8] = h;
}

// ---------------------------------------------------------------------------
// GEMM1: H = relu(A @ W1 + b1), A = agg [50000 x 128] fp32, W1 pre-packed f16.
// Block = 256 threads = 8 waves; wave w owns N-tile w (N=128 -> 8 tiles).
// ---------------------------------------------------------------------------
__global__ void __launch_bounds__(256)
gemm1_relu(const float* __restrict__ A, const _Float16* __restrict__ W1h,
           const float* __restrict__ b1, float* __restrict__ H) {
    __shared__ __align__(16) _Float16 aF[4][32][16];
    int lane   = threadIdx.x & 31;
    int wave   = threadIdx.x >> 5;
    int blockM = blockIdx.x;
    int  l15 = lane & 15;
    bool hi  = lane >= 16;

    stage_a_frags(A, blockM, threadIdx.x, aF);
    __syncthreads();

    v8f c = {};
#pragma unroll
    for (int kt = 0; kt < 4; ++kt) {
        v16h a = *(const v16h*)aF[kt][lane];
        v16h b = *(const v16h*)(W1h + ((size_t)(wave * 4 + kt) * 32 + lane) * 16);
        c = __builtin_amdgcn_wmma_f32_16x16x32_f16(false, a, false, b,
                                                   (short)0, c, false, false);
    }
    int n = wave * 16 + l15;
    float bias = b1[n];
#pragma unroll
    for (int r = 0; r < 8; ++r) {
        int m = blockM * 16 + (hi ? 8 + r : r);
        float v = c[r] + bias;
        H[(size_t)m * FEAT + n] = v > 0.0f ? v : 0.0f;
    }
}

// ---------------------------------------------------------------------------
// GEMM2 fused: logits = H @ W2 + b2 -> row softmax -> column-sum -> atomic fp.
// Block = 16 atom rows; 8 waves x 16 N-tiles each = 2048 columns.
// Logit tile kept entirely in VGPRs (16 x v8f per wave).
// ---------------------------------------------------------------------------
__global__ void __launch_bounds__(256)
gemm2_softmax_fp(const float* __restrict__ H, const _Float16* __restrict__ W2h,
                 const float* __restrict__ b2, float* __restrict__ fp) {
    __shared__ __align__(16) _Float16 aF[4][32][16];
    __shared__ float redbuf[8][16];
    int lane   = threadIdx.x & 31;
    int wave   = threadIdx.x >> 5;
    int blockM = blockIdx.x;
    int  l15 = lane & 15;
    bool hi  = lane >= 16;
    int  roff = hi ? 8 : 0;

    stage_a_frags(H, blockM, threadIdx.x, aF);
    __syncthreads();

    v16h a[4];
#pragma unroll
    for (int kt = 0; kt < 4; ++kt) a[kt] = *(const v16h*)aF[kt][lane];

    v8f acc[16];
#pragma unroll
    for (int i = 0; i < 16; ++i) {
        int nt = wave * 16 + i;
        // prefetch next N-tile's B fragments while this tile computes
        if (i < 15) {
            __builtin_prefetch(W2h + ((size_t)((nt + 1) * 4) * 32 + lane) * 16, 0, 1);
        }
        v8f c = {};
#pragma unroll
        for (int kt = 0; kt < 4; ++kt) {
            v16h b = *(const v16h*)(W2h + ((size_t)(nt * 4 + kt) * 32 + lane) * 16);
            c = __builtin_amdgcn_wmma_f32_16x16x32_f16(false, a[kt], false, b,
                                                       (short)0, c, false, false);
        }
        float bias = b2[nt * 16 + l15];
#pragma unroll
        for (int r = 0; r < 8; ++r) c[r] += bias;
        acc[i] = c;
    }

    // --- per-row max over this wave's 256 columns -------------------------
    float part[8];
#pragma unroll
    for (int r = 0; r < 8; ++r) {
        float m = acc[0][r];
#pragma unroll
        for (int i = 1; i < 16; ++i) m = fmaxf(m, acc[i][r]);
        part[r] = m;
    }
#pragma unroll
    for (int msk = 1; msk < 16; msk <<= 1) {
#pragma unroll
        for (int r = 0; r < 8; ++r)
            part[r] = fmaxf(part[r], __shfl_xor(part[r], msk, 32));
    }
    if (l15 == 0) {
#pragma unroll
        for (int r = 0; r < 8; ++r) redbuf[wave][roff + r] = part[r];
    }
    __syncthreads();
    float rowmax[8];
#pragma unroll
    for (int r = 0; r < 8; ++r) {
        float m = redbuf[0][roff + r];
#pragma unroll
        for (int w = 1; w < 8; ++w) m = fmaxf(m, redbuf[w][roff + r]);
        rowmax[r] = m;
    }
    __syncthreads();

    // --- exp + per-row sums ----------------------------------------------
#pragma unroll
    for (int r = 0; r < 8; ++r) part[r] = 0.0f;
#pragma unroll
    for (int i = 0; i < 16; ++i) {
#pragma unroll
        for (int r = 0; r < 8; ++r) {
            float e = __expf(acc[i][r] - rowmax[r]);
            acc[i][r] = e;
            part[r]  += e;
        }
    }
#pragma unroll
    for (int msk = 1; msk < 16; msk <<= 1) {
#pragma unroll
        for (int r = 0; r < 8; ++r)
            part[r] += __shfl_xor(part[r], msk, 32);
    }
    if (l15 == 0) {
#pragma unroll
        for (int r = 0; r < 8; ++r) redbuf[wave][roff + r] = part[r];
    }
    __syncthreads();
    float rinv[8];
#pragma unroll
    for (int r = 0; r < 8; ++r) {
        float s = 0.0f;
#pragma unroll
        for (int w = 0; w < 8; ++w) s += redbuf[w][roff + r];
        rinv[r] = 1.0f / s;
    }

    // --- column sums of probabilities, one atomic per column per block ----
#pragma unroll
    for (int i = 0; i < 16; ++i) {
        float cs = 0.0f;
#pragma unroll
        for (int r = 0; r < 8; ++r) cs += acc[i][r] * rinv[r];
        cs += __shfl_xor(cs, 16, 32);   // combine rows 0..7 with rows 8..15
        if (!hi) atomAddF(&fp[(wave * 16 + i) * 16 + l15], cs);
    }
}

// ---------------------------------------------------------------------------
extern "C" void kernel_launch(void* const* d_in, const int* in_sizes, int n_in,
                              void* d_out, int out_size, void* d_ws, size_t ws_size,
                              hipStream_t stream) {
    const float* atoms = (const float*)d_in[0];
    const float* W1    = (const float*)d_in[1];
    const float* b1    = (const float*)d_in[2];
    const float* W2    = (const float*)d_in[3];
    const float* b2    = (const float*)d_in[4];
    const int*   esrc  = (const int*)d_in[5];
    const int*   edst  = (const int*)d_in[6];
    float* fp = (float*)d_out;

    int nEdges  = in_sizes[5];
    int nAtoms  = in_sizes[0] / FEAT;   // 50000
    int mBlocks = nAtoms / 16;          // 3125 (exact)

    // workspace carve-out
    char* ws = (char*)d_ws;
    _Float16* W1h = (_Float16*)ws;  ws += (size_t)FEAT * FEAT   * sizeof(_Float16);
    _Float16* W2h = (_Float16*)ws;  ws += (size_t)FEAT * FP_LEN * sizeof(_Float16);
    ws = (char*)(((uintptr_t)ws + 255) & ~(uintptr_t)255);
    float* bufA = (float*)ws;       ws += (size_t)nAtoms * FEAT * sizeof(float);
    ws = (char*)(((uintptr_t)ws + 255) & ~(uintptr_t)255);
    float* bufB = (float*)ws;

    // one-time per launch: pack weights to f16 B-fragments, zero output
    {
        int tot1 = (FEAT / 32) * (FEAT / 16) * 32;     // 1024
        pack_weights<<<(tot1 + 255) / 256, 256, 0, stream>>>(W1, W1h, FEAT, FEAT);
        int tot2 = (FEAT / 32) * (FP_LEN / 16) * 32;   // 16384
        pack_weights<<<(tot2 + 255) / 256, 256, 0, stream>>>(W2, W2h, FEAT, FP_LEN);
        zero_kernel<<<(FP_LEN + 255) / 256, 256, 0, stream>>>(fp, FP_LEN);
    }

    const float* x = atoms;
    for (int step = 0; step < RADIUS; ++step) {
        int n4 = nAtoms * FEAT / 4;
        copy_f4<<<(n4 + 255) / 256, 256, 0, stream>>>((const float4*)x, (float4*)bufB, n4);

        long long totE = (long long)nEdges * 32;
        scatter_edges<<<(int)((totE + 255) / 256), 256, 0, stream>>>(x, esrc, edst, bufB, nEdges);

        gemm1_relu<<<mBlocks, 256, 0, stream>>>(bufB, W1h, b1, bufA);
        gemm2_softmax_fp<<<mBlocks, 256, 0, stream>>>(bufA, W2h, b2, fp);

        x = bufA;
    }
}
